// EncoderProcessorDecoder_87608742903948
// MI455X (gfx1250) — compile-verified, matrix operand order checked by reading, not run on hardware
//
#include <hip/hip_runtime.h>
#include <hip/hip_bf16.h>

#define H 128

typedef __attribute__((ext_vector_type(16))) __bf16 v16bf;
typedef __attribute__((ext_vector_type(8)))  float  v8f;

union AFrag { v16bf v; unsigned int u[8]; };
union CFrag { v8f v; float f[8]; };

// K index held by element e (0..15) of a lane in half 'hi' of a 16x32 bf16
// A fragment (ISA 7.12.2). B fragment uses the mirrored layout (lane = column).
__device__ __forceinline__ int wmma_k_of(int e, int hi) {
    return ((e < 8) ? e : (e + 8)) + 8 * hi;
}

// ---------------------------------------------------------------------------
// Pack fp32 row-major W[K][Nact] into WMMA B-fragment order as bf16.
// idx = ((nt*KT + kt)*32 + lane)*16 + e
//   k = kt*32 + wmma_k_of(e, lane>>4), n = nt*16 + (lane&15)  (zero-pad n>=Nact)
// ---------------------------------------------------------------------------
__global__ void pack_w_kernel(const float* __restrict__ W, __bf16* __restrict__ out,
                              int K, int Nact, int Npad)
{
    int total = K * Npad;
    int KT = K >> 5;
    for (int idx = blockIdx.x * blockDim.x + threadIdx.x; idx < total;
         idx += gridDim.x * blockDim.x) {
        int e    = idx & 15;
        int lane = (idx >> 4) & 31;
        int t    = idx >> 9;
        int kt   = t % KT;
        int nt   = t / KT;
        int k = kt * 32 + wmma_k_of(e, lane >> 4);
        int n = nt * 16 + (lane & 15);
        float v = (n < Nact) ? W[(size_t)k * Nact + n] : 0.0f;
        out[idx] = (__bf16)v;
    }
}

__global__ void zero_i32_kernel(int* __restrict__ p, int n)
{
    for (int i = blockIdx.x * blockDim.x + threadIdx.x; i < n;
         i += gridDim.x * blockDim.x) p[i] = 0;
}

// ---- CSR construction (once per launch; graph is static across steps) ----
__global__ void count_kernel(const int* __restrict__ rcv, int* __restrict__ counts, int E)
{
    for (int e = blockIdx.x * blockDim.x + threadIdx.x; e < E;
         e += gridDim.x * blockDim.x)
        atomicAdd(&counts[rcv[e]], 1);
}

// Single-block chunked exclusive scan: offsets[0..N], offsets[N] = E.
__global__ void __launch_bounds__(1024)
scan_offsets_kernel(const int* __restrict__ counts, int* __restrict__ offsets, int N)
{
    __shared__ int buf[1024];
    __shared__ int basebuf;
    int tid = threadIdx.x;
    if (tid == 0) basebuf = 0;
    __syncthreads();
    for (int start = 0; start < N; start += 1024) {
        int idx = start + tid;
        int v = (idx < N) ? counts[idx] : 0;
        buf[tid] = v;
        __syncthreads();
        for (int d = 1; d < 1024; d <<= 1) {      // inclusive Hillis-Steele
            int t = (tid >= d) ? buf[tid - d] : 0;
            __syncthreads();
            buf[tid] += t;
            __syncthreads();
        }
        int incl = buf[tid];
        int base = basebuf;
        if (idx < N) offsets[idx] = base + incl - v;   // exclusive
        __syncthreads();
        if (tid == 1023) basebuf = base + buf[1023];
        __syncthreads();
    }
    if (tid == 0) offsets[N] = basebuf;
}

__global__ void fill_csr_kernel(const int* __restrict__ snd, const int* __restrict__ rcv,
                                const int* __restrict__ offsets, int* __restrict__ cur,
                                int* __restrict__ srcIdx, int E)
{
    for (int e = blockIdx.x * blockDim.x + threadIdx.x; e < E;
         e += gridDim.x * blockDim.x) {
        int r = rcv[e];
        int pos = offsets[r] + atomicAdd(&cur[r], 1);
        srcIdx[pos] = snd[e];
    }
}

// ---------------------------------------------------------------------------
// Fused: agg = CSR-gather-sum(h_old); out = resid + LN(MLP(concat(h,agg)))
// One block = 16 rows, 4 waves; each wave owns 2 of the 8 16-col output tiles
// and gathers 4 of the 16 rows. offsets==null -> encoder mode (K1 = 128).
// h_old fits in the 192MB L2, so per-edge row gathers are mostly L2 hits.
// ---------------------------------------------------------------------------
__global__ void __launch_bounds__(128)
fused_mlp_ln_kernel(const float* __restrict__ X1,
                    const int* __restrict__ offsets, const int* __restrict__ srcIdx,
                    const __bf16* __restrict__ W1p, const float* __restrict__ b1,
                    const __bf16* __restrict__ W2p, const float* __restrict__ b2,
                    const float* __restrict__ g, const float* __restrict__ beta,
                    const float* __restrict__ resid, float* __restrict__ out,
                    int N, int K1)
{
    __shared__ __bf16 sA[16 * 256];
    __shared__ __bf16 sMid[16 * 128];
    __shared__ float  sOut[16 * 128];
    __shared__ float  sMu[16], sRs[16];

    const int tid  = threadIdx.x;
    const int lane = tid & 31;
    const int wave = tid >> 5;
    const int m16  = lane & 15;
    const int hi   = lane >> 4;
    const int row0 = blockIdx.x * 16;
    const int KT1  = K1 >> 5;

    // Stage node features: X1 rows -> sA cols [0, H)  (f32 -> bf16)
    for (int i = tid; i < (16 * H) >> 2; i += 128) {
        int rr = (i << 2) >> 7;
        int cc = (i << 2) & 127;
        int rsrc = row0 + rr; if (rsrc >= N) rsrc = N - 1;
        float4 v = *(const float4*)(X1 + (size_t)rsrc * H + cc);
        __bf16* d = &sA[rr * K1 + cc];
        d[0] = (__bf16)v.x; d[1] = (__bf16)v.y;
        d[2] = (__bf16)v.z; d[3] = (__bf16)v.w;
    }

    // Fused CSR gather-sum: wave w accumulates rows 4w..4w+3 -> sA cols [H, 2H)
    if (offsets != nullptr) {
        for (int q = 0; q < 4; ++q) {
            int rr  = wave * 4 + q;
            int row = row0 + rr;
            float4 acc = make_float4(0.f, 0.f, 0.f, 0.f);
            if (row < N) {
                int beg = offsets[row], end = offsets[row + 1];
                for (int e = beg; e < end; ++e) {
                    int s = srcIdx[e];   // wave-uniform -> scalar load
                    float4 v = ((const float4*)(X1 + (size_t)s * H))[lane];
                    acc.x += v.x; acc.y += v.y; acc.z += v.z; acc.w += v.w;
                }
            }
            __bf16* d = &sA[rr * 256 + H + lane * 4];
            d[0] = (__bf16)acc.x; d[1] = (__bf16)acc.y;
            d[2] = (__bf16)acc.z; d[3] = (__bf16)acc.w;
        }
    }
    __syncthreads();

    // GEMM1 + bias + ReLU -> sMid (bf16)
    for (int j = 0; j < 2; ++j) {
        int nt = wave * 2 + j;
        CFrag acc;
        float bv = b1[nt * 16 + m16];
        #pragma unroll
        for (int r = 0; r < 8; ++r) acc.f[r] = bv;
        const __bf16* wb = W1p + ((size_t)nt * KT1) * 512 + lane * 16;
        const unsigned int* arow = (const unsigned int*)&sA[m16 * K1];
        for (int kt = 0; kt < KT1; ++kt) {
            AFrag a, b;
            #pragma unroll
            for (int p = 0; p < 8; ++p) {
                int k0 = kt * 32 + wmma_k_of(2 * p, hi);
                a.u[p] = arow[k0 >> 1];
            }
            const uint4* bp = (const uint4*)(wb + (size_t)kt * 512);
            uint4 q0 = bp[0], q1 = bp[1];
            b.u[0] = q0.x; b.u[1] = q0.y; b.u[2] = q0.z; b.u[3] = q0.w;
            b.u[4] = q1.x; b.u[5] = q1.y; b.u[6] = q1.z; b.u[7] = q1.w;
            acc.v = __builtin_amdgcn_wmma_f32_16x16x32_bf16(
                false, a.v, false, b.v, (short)0, acc.v, false, false);
        }
        #pragma unroll
        for (int r = 0; r < 8; ++r) {
            float vv = acc.f[r] > 0.0f ? acc.f[r] : 0.0f;
            int mm = r + 8 * hi;
            sMid[mm * 128 + nt * 16 + m16] = (__bf16)vv;
        }
    }
    __syncthreads();

    // GEMM2 + bias -> sOut (f32)
    for (int j = 0; j < 2; ++j) {
        int nt = wave * 2 + j;
        CFrag acc;
        float bv = b2[nt * 16 + m16];
        #pragma unroll
        for (int r = 0; r < 8; ++r) acc.f[r] = bv;
        const __bf16* wb = W2p + ((size_t)nt * 4) * 512 + lane * 16;
        const unsigned int* arow = (const unsigned int*)&sMid[m16 * 128];
        #pragma unroll
        for (int kt = 0; kt < 4; ++kt) {
            AFrag a, b;
            #pragma unroll
            for (int p = 0; p < 8; ++p) {
                int k0 = kt * 32 + wmma_k_of(2 * p, hi);
                a.u[p] = arow[k0 >> 1];
            }
            const uint4* bp = (const uint4*)(wb + (size_t)kt * 512);
            uint4 q0 = bp[0], q1 = bp[1];
            b.u[0] = q0.x; b.u[1] = q0.y; b.u[2] = q0.z; b.u[3] = q0.w;
            b.u[4] = q1.x; b.u[5] = q1.y; b.u[6] = q1.z; b.u[7] = q1.w;
            acc.v = __builtin_amdgcn_wmma_f32_16x16x32_bf16(
                false, a.v, false, b.v, (short)0, acc.v, false, false);
        }
        #pragma unroll
        for (int r = 0; r < 8; ++r) {
            int mm = r + 8 * hi;
            sOut[mm * 128 + nt * 16 + m16] = acc.f[r];
        }
    }
    __syncthreads();

    // Row-wise LayerNorm stats
    if (tid < 16) {
        float mu = 0.0f;
        for (int c = 0; c < 128; ++c) mu += sOut[tid * 128 + c];
        mu *= (1.0f / 128.0f);
        float var = 0.0f;
        for (int c = 0; c < 128; ++c) {
            float d = sOut[tid * 128 + c] - mu;
            var += d * d;
        }
        var *= (1.0f / 128.0f);
        sMu[tid] = mu;
        sRs[tid] = rsqrtf(var + 1e-5f);
    }
    __syncthreads();

    for (int i = tid; i < 16 * 128; i += 128) {
        int rr = i >> 7, cc = i & 127;
        int row = row0 + rr;
        if (row < N) {
            float v = (sOut[i] - sMu[rr]) * sRs[rr] * g[cc] + beta[cc];
            if (resid) v += resid[(size_t)row * H + cc];
            out[(size_t)row * H + cc] = v;
        }
    }
}

// ---------------------------------------------------------------------------
// Decoder: ReLU(h @ W1 + b1) @ W2 + b2, OUT = 3 (W2 packed zero-padded to 16).
// ---------------------------------------------------------------------------
__global__ void __launch_bounds__(128)
decoder_kernel(const float* __restrict__ Xh,
               const __bf16* __restrict__ W1p, const float* __restrict__ b1,
               const __bf16* __restrict__ W2p, const float* __restrict__ b2,
               float* __restrict__ out, int N)
{
    __shared__ __bf16 sA[16 * 128];
    __shared__ __bf16 sMid[16 * 128];

    const int tid  = threadIdx.x;
    const int lane = tid & 31;
    const int wave = tid >> 5;
    const int m16  = lane & 15;
    const int hi   = lane >> 4;
    const int row0 = blockIdx.x * 16;

    for (int i = tid; i < (16 * 128) >> 2; i += 128) {
        int rr = (i << 2) >> 7;
        int cc = (i << 2) & 127;
        int rsrc = row0 + rr; if (rsrc >= N) rsrc = N - 1;
        float4 v = *(const float4*)(Xh + (size_t)rsrc * H + cc);
        __bf16* d = &sA[rr * 128 + cc];
        d[0] = (__bf16)v.x; d[1] = (__bf16)v.y;
        d[2] = (__bf16)v.z; d[3] = (__bf16)v.w;
    }
    __syncthreads();

    for (int j = 0; j < 2; ++j) {
        int nt = wave * 2 + j;
        CFrag acc;
        float bv = b1[nt * 16 + m16];
        #pragma unroll
        for (int r = 0; r < 8; ++r) acc.f[r] = bv;
        const __bf16* wb = W1p + ((size_t)nt * 4) * 512 + lane * 16;
        const unsigned int* arow = (const unsigned int*)&sA[m16 * 128];
        #pragma unroll
        for (int kt = 0; kt < 4; ++kt) {
            AFrag a, b;
            #pragma unroll
            for (int p = 0; p < 8; ++p) {
                int k0 = kt * 32 + wmma_k_of(2 * p, hi);
                a.u[p] = arow[k0 >> 1];
            }
            const uint4* bp = (const uint4*)(wb + (size_t)kt * 512);
            uint4 q0 = bp[0], q1 = bp[1];
            b.u[0] = q0.x; b.u[1] = q0.y; b.u[2] = q0.z; b.u[3] = q0.w;
            b.u[4] = q1.x; b.u[5] = q1.y; b.u[6] = q1.z; b.u[7] = q1.w;
            acc.v = __builtin_amdgcn_wmma_f32_16x16x32_bf16(
                false, a.v, false, b.v, (short)0, acc.v, false, false);
        }
        #pragma unroll
        for (int r = 0; r < 8; ++r) {
            float vv = acc.f[r] > 0.0f ? acc.f[r] : 0.0f;
            int mm = r + 8 * hi;
            sMid[mm * 128 + nt * 16 + m16] = (__bf16)vv;
        }
    }
    __syncthreads();

    if (wave == 0) {   // single output tile; EXEC full within the wave
        CFrag acc;
        float bv = (m16 < 3) ? b2[m16] : 0.0f;
        #pragma unroll
        for (int r = 0; r < 8; ++r) acc.f[r] = bv;
        const __bf16* wb = W2p + lane * 16;
        const unsigned int* arow = (const unsigned int*)&sMid[m16 * 128];
        #pragma unroll
        for (int kt = 0; kt < 4; ++kt) {
            AFrag a, b;
            #pragma unroll
            for (int p = 0; p < 8; ++p) {
                int k0 = kt * 32 + wmma_k_of(2 * p, hi);
                a.u[p] = arow[k0 >> 1];
            }
            const uint4* bp = (const uint4*)(wb + (size_t)kt * 512);
            uint4 q0 = bp[0], q1 = bp[1];
            b.u[0] = q0.x; b.u[1] = q0.y; b.u[2] = q0.z; b.u[3] = q0.w;
            b.u[4] = q1.x; b.u[5] = q1.y; b.u[6] = q1.z; b.u[7] = q1.w;
            acc.v = __builtin_amdgcn_wmma_f32_16x16x32_bf16(
                false, a.v, false, b.v, (short)0, acc.v, false, false);
        }
        #pragma unroll
        for (int r = 0; r < 8; ++r) {
            int row = row0 + r + 8 * hi;
            if (m16 < 3 && row < N) out[(size_t)row * 3 + m16] = acc.f[r];
        }
    }
}

// ---------------------------------------------------------------------------
extern "C" void kernel_launch(void* const* d_in, const int* in_sizes, int n_in,
                              void* d_out, int out_size, void* d_ws, size_t ws_size,
                              hipStream_t stream)
{
    (void)n_in; (void)out_size; (void)ws_size;
    const float* x         = (const float*)d_in[0];
    const int*   senders   = (const int*)d_in[1];
    const int*   receivers = (const int*)d_in[2];
    const float* enc_w1    = (const float*)d_in[3];
    const float* enc_b1    = (const float*)d_in[4];
    const float* enc_w2    = (const float*)d_in[5];
    const float* enc_b2    = (const float*)d_in[6];
    const float* enc_g     = (const float*)d_in[7];
    const float* enc_beta  = (const float*)d_in[8];
    const float* pw1       = (const float*)d_in[9];
    const float* pb1       = (const float*)d_in[10];
    const float* pw2       = (const float*)d_in[11];
    const float* pb2       = (const float*)d_in[12];
    const float* pg        = (const float*)d_in[13];
    const float* pbeta     = (const float*)d_in[14];
    const float* dec_w1    = (const float*)d_in[15];
    const float* dec_b1    = (const float*)d_in[16];
    const float* dec_w2    = (const float*)d_in[17];
    const float* dec_b2    = (const float*)d_in[18];

    const int N = in_sizes[0] / H;
    const int E = in_sizes[1];
    const int S = in_sizes[9] / (2 * H * H);

    char* ws = (char*)d_ws;
    size_t off = 0;
    auto take = [&](size_t bytes) -> char* {
        char* p = ws + off;
        off += (bytes + 255) & ~(size_t)255;
        return p;
    };
    float*  h0      = (float*)take((size_t)N * H * sizeof(float));
    float*  h1      = (float*)take((size_t)N * H * sizeof(float));
    int*    counts  = (int*)take((size_t)N * sizeof(int));        // reused as cursor
    int*    offsets = (int*)take((size_t)(N + 1) * sizeof(int));
    int*    srcIdx  = (int*)take((size_t)E * sizeof(int));
    __bf16* enc_w1p = (__bf16*)take((size_t)H * H * 2);
    __bf16* enc_w2p = (__bf16*)take((size_t)H * H * 2);
    __bf16* pw1p    = (__bf16*)take((size_t)S * 2 * H * H * 2);
    __bf16* pw2p    = (__bf16*)take((size_t)S * H * H * 2);
    __bf16* dec_w1p = (__bf16*)take((size_t)H * H * 2);
    __bf16* dec_w2p = (__bf16*)take((size_t)H * 16 * 2);

    // Pack all weights into WMMA bf16 fragment order (tiny; L2-resident).
    pack_w_kernel<<<(H * H + 255) / 256, 256, 0, stream>>>(enc_w1, enc_w1p, H, H, H);
    pack_w_kernel<<<(H * H + 255) / 256, 256, 0, stream>>>(enc_w2, enc_w2p, H, H, H);
    pack_w_kernel<<<(H * H + 255) / 256, 256, 0, stream>>>(dec_w1, dec_w1p, H, H, H);
    pack_w_kernel<<<(H * 16 + 255) / 256, 256, 0, stream>>>(dec_w2, dec_w2p, H, 3, 16);
    for (int s = 0; s < S; ++s) {
        pack_w_kernel<<<(2 * H * H + 255) / 256, 256, 0, stream>>>(
            pw1 + (size_t)s * 2 * H * H, pw1p + (size_t)s * 2 * H * H, 2 * H, H, H);
        pack_w_kernel<<<(H * H + 255) / 256, 256, 0, stream>>>(
            pw2 + (size_t)s * H * H, pw2p + (size_t)s * H * H, H, H, H);
    }

    // Build CSR once (graph static across all S steps): no per-step atomics.
    zero_i32_kernel<<<(N + 255) / 256, 256, 0, stream>>>(counts, N);
    count_kernel<<<(E + 255) / 256, 256, 0, stream>>>(receivers, counts, E);
    scan_offsets_kernel<<<1, 1024, 0, stream>>>(counts, offsets, N);
    zero_i32_kernel<<<(N + 255) / 256, 256, 0, stream>>>(counts, N);
    fill_csr_kernel<<<(E + 255) / 256, 256, 0, stream>>>(senders, receivers, offsets,
                                                         counts, srcIdx, E);

    const int nblk = (N + 15) / 16;

    // Encoder: h0 = LN(MLP(x))
    fused_mlp_ln_kernel<<<nblk, 128, 0, stream>>>(
        x, nullptr, nullptr, enc_w1p, enc_b1, enc_w2p, enc_b2, enc_g, enc_beta,
        nullptr, h0, N, H);

    // Processor steps: ping-pong h (gather reads other rows -> no in-place).
    float* hs = h0;
    float* hd = h1;
    for (int s = 0; s < S; ++s) {
        fused_mlp_ln_kernel<<<nblk, 128, 0, stream>>>(
            hs, offsets, srcIdx,
            pw1p + (size_t)s * 2 * H * H, pb1 + (size_t)s * H,
            pw2p + (size_t)s * H * H,     pb2 + (size_t)s * H,
            pg + (size_t)s * H, pbeta + (size_t)s * H,
            hs, hd, N, 2 * H);
        float* t = hs; hs = hd; hd = t;
    }

    // Decoder
    decoder_kernel<<<nblk, 128, 0, stream>>>(hs, dec_w1p, dec_b1, dec_w2p, dec_b2,
                                             (float*)d_out, N);
}
